// NeuralGRDE_83519934038520
// MI455X (gfx1250) — matrix operand order repaired and checked by reading, not compiled, not run on hardware
//
#include <hip/hip_runtime.h>

// ---------------------------------------------------------------------------
// Neural CDE (RK4) fused persistent kernel for MI455X (gfx1250, wave32, WMMA)
//   B=128, N=325, T=24, C=2, H=256, HORIZON=12, OUT=1
//   rows = B*N = 41600, each block integrates 16 rows for all 23 steps.
// ---------------------------------------------------------------------------

typedef __attribute__((ext_vector_type(16))) __bf16 v16bf;
typedef __attribute__((ext_vector_type(8)))  float  v8f;

#define BB    128
#define NN    325
#define TT    24
#define CC    2
#define HH    256
#define HC    512          // H*C
#define HOR   12
#define RPB   16           // rows per block
#define ZP    260          // padded row pitch (floats) -> conflict-free LDS
#define YP    20           // y-staging column pitch (floats), b128-conflict-free

// --------------------------------------------------------------------------
// Prep: swizzle Wg (f32 [256][512] row-major, Wg[k][n]) into bf16 B-fragments
// for v_wmma_f32_16x16x32_bf16 B in the contiguous k order also used for A (wave32):
//   fragment f = ntile*8 + ktile ; per lane l: n = ntile*16 + (l&15),
//   K = ktile*32 + (l<16 ? 0 : 16) + e , e = 0..15 contiguous per lane.
// --------------------------------------------------------------------------
__global__ void grde_prep_wg_bf16(const float* __restrict__ Wg,
                                  __bf16* __restrict__ wsB) {
  int t = blockIdx.x * blockDim.x + threadIdx.x;     // 8192 threads
  if (t >= 256 * 32) return;
  int f  = t >> 5;
  int l  = t & 31;
  int nt = f >> 3;
  int kt = f & 7;
  int n     = nt * 16 + (l & 15);
  int kbase = kt * 32 + ((l & 16) ? 16 : 0);
#pragma unroll
  for (int e = 0; e < 16; ++e) {
    int k = kbase + e;
    wsB[f * 512 + l * 16 + e] = (__bf16)Wg[k * HC + n];
  }
}

// --------------------------------------------------------------------------
// Main persistent kernel: one block = 16 rows, full 23-step RK4 integration.
// --------------------------------------------------------------------------
__global__ __launch_bounds__(256, 1) void grde_kernel(
    const float* __restrict__ times, const float* __restrict__ coeffs,
    const float* __restrict__ Wz,    const float* __restrict__ bz,
    const __bf16* __restrict__ wsB,  const float* __restrict__ bg,
    const float* __restrict__ Wend,  const float* __restrict__ bend,
    float* __restrict__ out) {
  __shared__ __align__(16) __bf16 sB[256 * 512];     // 256 KB swizzled Wg
  __shared__ __align__(16) float  sZ[RPB * ZP];      // z state (f32)
  __shared__ __align__(16) float  sK[RPB * ZP];      // current stage k
  __shared__ __align__(16) float  sY[8 * 16 * YP];   // per-wave y staging
  __shared__ float sDx[RPB * CC];

  const int tid   = threadIdx.x;
  const int lane  = tid & 31;
  const int wave  = tid >> 5;
  const int mrow  = lane & 15;   // A-row within tile / y-tile column index
  const int half  = lane >> 4;
  const int rbase = blockIdx.x * RPB;

  // ---- async-copy swizzled Wg into LDS (CDNA5 global->LDS async path) ----
  {
    unsigned lds_base = (unsigned)(uintptr_t)(&sB[0]);   // wave-relative LDS addr
    unsigned long long gsrc = (unsigned long long)(uintptr_t)wsB;
    for (int i = tid; i < (256 * 512 * 2) / 16; i += 256) {
      unsigned           d = lds_base + (unsigned)i * 16u;
      unsigned long long s = gsrc + (unsigned long long)i * 16ull;
      asm volatile("global_load_async_to_lds_b128 %0, %1, off"
                   :: "v"(d), "v"(s) : "memory");
    }
    asm volatile("s_wait_asynccnt 0x0" ::: "memory");
  }
  // ---- z0 = coeffs[...,0,:] @ Wz + bz ----
  for (int idx = tid; idx < RPB * HH; idx += 256) {
    int m = idx >> 8, h = idx & 255;
    int row  = rbase + m;
    float c0 = coeffs[(row * TT + 0) * CC + 0];
    float c1 = coeffs[(row * TT + 0) * CC + 1];
    sZ[m * ZP + h] = c0 * Wz[h] + c1 * Wz[HH + h] + bz[h];
  }

  // ---- loop-invariant per-lane constants ----
  float biasj[4];
#pragma unroll
  for (int j = 0; j < 4; ++j) biasj[j] = bg[(wave * 4 + j) * 16 + mrow];
  float* const yb = &sY[wave * 16 * YP];

  __syncthreads();

  for (int t = 0; t < TT - 1; ++t) {
    float dt = times[t + 1] - times[t];
    if (tid < RPB * CC) {
      int m = tid >> 1, c = tid & 1;
      int row = rbase + m;
      sDx[tid] = (coeffs[(row * TT + t + 1) * CC + c] -
                  coeffs[(row * TT + t) * CC + c]) / dt;
    }
    __syncthreads();
    const float dx0 = sDx[mrow * 2 + 0];
    const float dx1 = sDx[mrow * 2 + 1];

    float accR[16];    // RK accumulator: (row=mrow, h = nt*8 + half*4 + q)

    for (int s = 0; s < 4; ++s) {
      const float a   = (s == 0) ? 0.0f : ((s == 3) ? 1.0f : 0.5f);
      const float w   = (s == 1 || s == 2) ? 2.0f : 1.0f;
      const float adt = a * dt;

      // ---- build 8 bf16 A fragments of (z + a*dt*k) for our 16 rows ----
      v16bf A[8];
#pragma unroll
      for (int kt = 0; kt < 8; ++kt) {
        int k0 = kt * 32 + (half ? 16 : 0);   // e = 0..7
        int k1 = kt * 32 + (half ? 24 : 8);   // e = 8..15
        const float* zr = &sZ[mrow * ZP];
        float4 z0 = *reinterpret_cast<const float4*>(zr + k0);
        float4 z1 = *reinterpret_cast<const float4*>(zr + k0 + 4);
        float4 z2 = *reinterpret_cast<const float4*>(zr + k1);
        float4 z3 = *reinterpret_cast<const float4*>(zr + k1 + 4);
        float v[16] = {z0.x, z0.y, z0.z, z0.w, z1.x, z1.y, z1.z, z1.w,
                       z2.x, z2.y, z2.z, z2.w, z3.x, z3.y, z3.z, z3.w};
        if (s > 0) {
          const float* kr = &sK[mrow * ZP];
          float4 q0 = *reinterpret_cast<const float4*>(kr + k0);
          float4 q1 = *reinterpret_cast<const float4*>(kr + k0 + 4);
          float4 q2 = *reinterpret_cast<const float4*>(kr + k1);
          float4 q3 = *reinterpret_cast<const float4*>(kr + k1 + 4);
          float q[16] = {q0.x, q0.y, q0.z, q0.w, q1.x, q1.y, q1.z, q1.w,
                         q2.x, q2.y, q2.z, q2.w, q3.x, q3.y, q3.z, q3.w};
#pragma unroll
          for (int i = 0; i < 16; ++i) v[i] += adt * q[i];
        }
#pragma unroll
        for (int i = 0; i < 16; ++i) A[kt][i] = (__bf16)v[i];
      }
      __syncthreads();  // all waves consumed sK/sZ; safe to overwrite sK

      // ---- each wave covers 4 of the 32 N-tiles of y = tanh(z@Wg+bg) ----
#pragma unroll
      for (int j = 0; j < 4; ++j) {
        const int nt = wave * 4 + j;

        // preload all 8 B fragments so DS latency overlaps the WMMA chain
        v16bf Bf[8];
#pragma unroll
        for (int kt = 0; kt < 8; ++kt)
          Bf[kt] = *reinterpret_cast<const v16bf*>(
              &sB[(nt * 8 + kt) * 512 + lane * 16]);

        v8f acc = {};
#pragma unroll
        for (int kt = 0; kt < 8; ++kt)
          acc = __builtin_amdgcn_wmma_f32_16x16x32_bf16(
              false, A[kt], false, Bf[kt], (short)0, acc, false, false);

        // bias + native tanh: consecutive independent v_tanh ops cover each
        // other's TRANS hazard; single trailing v_nop covers the last one.
        float y0 = acc[0] + biasj[j], y1 = acc[1] + biasj[j];
        float y2 = acc[2] + biasj[j], y3 = acc[3] + biasj[j];
        float y4 = acc[4] + biasj[j], y5 = acc[5] + biasj[j];
        float y6 = acc[6] + biasj[j], y7 = acc[7] + biasj[j];
        asm volatile(
            "v_tanh_f32 %0, %0\n\t"
            "v_tanh_f32 %1, %1\n\t"
            "v_tanh_f32 %2, %2\n\t"
            "v_tanh_f32 %3, %3\n\t"
            "v_tanh_f32 %4, %4\n\t"
            "v_tanh_f32 %5, %5\n\t"
            "v_tanh_f32 %6, %6\n\t"
            "v_tanh_f32 %7, %7\n\t"
            "v_nop"
            : "+v"(y0), "+v"(y1), "+v"(y2), "+v"(y3),
              "+v"(y4), "+v"(y5), "+v"(y6), "+v"(y7));

        // stage y column-major: yb[col*YP + row]; our col = mrow,
        // rows half*8 .. half*8+7 are contiguous -> two b128 stores.
        {
          float4 lo = {y0, y1, y2, y3};
          float4 hi = {y4, y5, y6, y7};
          float4* p = reinterpret_cast<float4*>(&yb[mrow * YP + half * 8]);
          p[0] = lo;
          p[1] = hi;
        }
        // contract even/odd y columns with dX (C = 2); DS ops of one wave
        // execute in order, so the staged y is visible to these reads.
#pragma unroll
        for (int q = 0; q < 4; ++q) {
          int   hl = half * 4 + q;
          float e0 = yb[(2 * hl + 0) * YP + mrow];
          float e1 = yb[(2 * hl + 1) * YP + mrow];
          float kv = e0 * dx0 + e1 * dx1;
          if (s < 3) sK[mrow * ZP + nt * 8 + hl] = kv;  // k4 feeds no stage
          int r = j * 4 + q;
          accR[r] = (s == 0) ? kv : accR[r] + w * kv;
        }
      }
      __syncthreads();  // sK complete for next stage's A build
    }

    // ---- z += dt/6 * (k1 + 2k2 + 2k3 + k4), per-lane owned (row,h) ----
    float c6 = dt * (1.0f / 6.0f);
#pragma unroll
    for (int j = 0; j < 4; ++j) {
#pragma unroll
      for (int q = 0; q < 4; ++q) {
        int h = (wave * 4 + j) * 8 + half * 4 + q;
        sZ[mrow * ZP + h] += c6 * accR[j * 4 + q];
      }
    }
    __syncthreads();
  }

  // ---- epilogue: out[b, o, n] = z[row] . Wend[o] + bend[o] ----
  if (tid < RPB * HOR) {
    int m = tid / HOR, o = tid % HOR;
    int row = rbase + m;
    int b = row / NN, n = row % NN;
    float sum = bend[o];
#pragma unroll 4
    for (int h = 0; h < HH; ++h) sum += sZ[m * ZP + h] * Wend[o * HH + h];
    out[(b * HOR + o) * NN + n] = sum;
  }
}

// --------------------------------------------------------------------------
extern "C" void kernel_launch(void* const* d_in, const int* in_sizes, int n_in,
                              void* d_out, int out_size, void* d_ws,
                              size_t ws_size, hipStream_t stream) {
  const float* times  = (const float*)d_in[0];
  const float* coeffs = (const float*)d_in[1];
  const float* Wz     = (const float*)d_in[2];
  const float* bz     = (const float*)d_in[3];
  const float* Wg     = (const float*)d_in[4];
  const float* bg     = (const float*)d_in[5];
  const float* Wend   = (const float*)d_in[6];
  const float* bend   = (const float*)d_in[7];
  float*  out = (float*)d_out;
  __bf16* wsB = (__bf16*)d_ws;  // 256 KB swizzled bf16 Wg image

  grde_prep_wg_bf16<<<32, 256, 0, stream>>>(Wg, wsB);
  grde_kernel<<<(BB * NN) / RPB, 256, 0, stream>>>(
      times, coeffs, Wz, bz, wsB, bg, Wend, bend, out);
}
